// GFormer_21792664059976
// MI455X (gfx1250) — matrix operand order repaired and checked by reading, not compiled, not run on hardware
//
#include <hip/hip_runtime.h>
#include <hip/hip_bf16.h>

// ---------------------------------------------------------------------------
// GFormer block on MI455X (gfx1250): BF16 WMMA pipeline, f32 accumulate.
// All GEMM B operands stored K-major so both fragment loads are contiguous
// 32B/lane (-> global_load_b128 pairs). Rectangular GEMMs stage the shared
// 16xK A panel in LDS via global_load_async_to_lds_b128 (+ s_wait_asynccnt).
// ---------------------------------------------------------------------------

typedef __attribute__((ext_vector_type(16))) __bf16 v16bf;
typedef __attribute__((ext_vector_type(8)))  float  v8f;
typedef unsigned short u16;

#define B_    8
#define C_    256
#define N_    768
#define T_    12
#define P_    (B_ * N_ * T_)      /* 73728 positions */
#define SCALE_ 0.125f             /* 64^-0.5 */

// ---- bf16 helpers ---------------------------------------------------------
__device__ __forceinline__ u16 f2bf(float f) {
  unsigned u = __builtin_bit_cast(unsigned, f);
  unsigned r = u + 0x7FFFu + ((u >> 16) & 1u);   // round-to-nearest-even
  return (u16)(r >> 16);
}
__device__ __forceinline__ float bf2f(u16 v) {
  unsigned u = ((unsigned)v) << 16;
  return __builtin_bit_cast(float, u);
}
__device__ __forceinline__ __bf16 bfu(u16 v) { return __builtin_bit_cast(__bf16, v); }

// A-matrix 16x32 bf16 per-lane K map (contiguous: k = 16h + e, matches B).
__device__ __forceinline__ int kmapA(int e, int h) {
  return h * 16 + e;
}

// A fragment: row-major A; base at (m0 row, k-col 0); lane row r.
__device__ __forceinline__ v16bf load_frag_a(const u16* base, long long rowStride,
                                             int r, int h, int k0) {
  v16bf f;
  const u16* p = base + (long long)r * rowStride + k0;
#pragma unroll
  for (int e = 0; e < 16; ++e) f[e] = bfu(p[kmapA(e, h)]);
  return f;
}

// B fragment from K-major storage BT[N][K]: lane column c reads 16 contiguous K.
__device__ __forceinline__ v16bf load_frag_bt(const u16* rowBase, long long rowStride,
                                              int c, int h, int k0) {
  v16bf f;
  const u16* p = rowBase + (long long)c * rowStride + k0 + h * 16;
#pragma unroll
  for (int e = 0; e < 16; ++e) f[e] = bfu(p[e]);
  return f;
}

__device__ __forceinline__ v8f wmma_bf16(v16bf a, v16bf b, v8f c) {
  return __builtin_amdgcn_wmma_f32_16x16x32_bf16(false, a, false, b, (short)0, c,
                                                 false, false);
}

// ---------------------------------------------------------------------------
// K1: pack weights -> bf16, K-major (PyTorch [out][in] is already K-major)
// ---------------------------------------------------------------------------
__global__ void pack_weights(const float* Wnq, const float* Wnk, const float* Wtq,
                             const float* Wtk, const float* bnq, const float* bnk,
                             const float* btq, const float* btk, const float* v_w,
                             const float* v_b, const float* fcv_w, const float* fcv_b,
                             const float* proj_w, const float* f1w, const float* f2w,
                             u16* Wp1T, u16* WprojT, u16* Wf1T, u16* Wf2T,
                             float* bias448) {
  int i = blockIdx.x * blockDim.x + threadIdx.x;
  if (i < 448 * 256) {                       // Wp1T[j][c], j = fused out-channel
    int j = i / 256, c = i % 256;
    float v;
    if (j < 64)       v = Wnq[c * 64 + j];
    else if (j < 128) v = Wnk[c * 64 + (j - 64)];
    else if (j < 192) v = Wtq[c * 64 + (j - 128)];
    else if (j < 256) v = Wtk[c * 64 + (j - 192)];
    else if (j < 320) v = v_w[(j - 256) * 256 + c];
    else              v = fcv_w[(j - 320) * 256 + c];
    Wp1T[i] = f2bf(v);
  }
  if (i < 448) {
    float v;
    if (i < 64)       v = bnq[i];
    else if (i < 128) v = bnk[i - 64];
    else if (i < 192) v = btq[i - 128];
    else if (i < 256) v = btk[i - 192];
    else if (i < 320) v = v_b[i - 256];
    else              v = fcv_b[i - 320];
    bias448[i] = v;
  }
  if (i < 256 * 128)  WprojT[i] = f2bf(proj_w[i]);   // [256][128] K-major already
  if (i < 1024 * 256) Wf1T[i]   = f2bf(f1w[i]);      // [1024][256]
  if (i < 256 * 512)  Wf2T[i]   = f2bf(f2w[i]);      // [256][512]
}

// ---------------------------------------------------------------------------
// LayerNorms
// ---------------------------------------------------------------------------
__global__ void ln1_kernel(const float* __restrict__ x, const float* w,
                           const float* b, u16* __restrict__ out) {
  int p = blockIdx.x;
  int c = threadIdx.x;                       // 256
  int bb = p / (N_ * T_);
  int rem = p % (N_ * T_);
  int n = rem / T_, t = rem % T_;
  float v = x[(((long long)bb * C_ + c) * N_ + n) * T_ + t];
  __shared__ float s1[256], s2[256];
  s1[c] = v; s2[c] = v * v;
  __syncthreads();
  for (int st = 128; st > 0; st >>= 1) {
    if (c < st) { s1[c] += s1[c + st]; s2[c] += s2[c + st]; }
    __syncthreads();
  }
  float mean = s1[0] * (1.0f / C_);
  float var = s2[0] * (1.0f / C_) - mean * mean;
  float inv = rsqrtf(var + 1e-6f);
  out[(long long)p * C_ + c] = f2bf(w[c] * (v - mean) * inv + b[c]);
}

__global__ void ln2_kernel(const float* __restrict__ xr, const float* w,
                           const float* b, u16* __restrict__ out) {
  int p = blockIdx.x;
  int c = threadIdx.x;
  float v = xr[(long long)p * C_ + c];
  __shared__ float s1[256], s2[256];
  s1[c] = v; s2[c] = v * v;
  __syncthreads();
  for (int st = 128; st > 0; st >>= 1) {
    if (c < st) { s1[c] += s1[c + st]; s2[c] += s2[c + st]; }
    __syncthreads();
  }
  float mean = s1[0] * (1.0f / C_);
  float var = s2[0] * (1.0f / C_) - mean * mean;
  float inv = rsqrtf(var + 1e-6f);
  out[(long long)p * C_ + c] = f2bf(w[c] * (v - mean) * inv + b[c]);
}

// ---------------------------------------------------------------------------
// Generic panel GEMM: C[16 x 16*NTILES] = A[16 x K] * BT[NTILES*16][K]^T
// A panel staged once into LDS with async copies; 8 waves/block on N tiles.
// ---------------------------------------------------------------------------
struct EpiProjAll {
  const float* bias; u16* QK; u16* Vbf; u16* VbfT; float* FCV;
  __device__ void operator()(long long m, int col, float val) const {
    val += bias[col];
    if (col < 256) {
      QK[m * 256 + col] = f2bf(val);
    } else if (col < 320) {
      int c = col - 256;
      u16 bv = f2bf(val);
      Vbf[m * 64 + c] = bv;                               // [p][64] for time attn
      int b = (int)(m / (N_ * T_));
      int rem = (int)(m % (N_ * T_));
      int n = rem / T_, t = rem % T_;
      VbfT[(((long long)b * T_ + t) * 64 + c) * N_ + n] = bv;  // slice-K-major
    } else {
      FCV[m * 128 + (col - 320)] = val;
    }
  }
};
struct EpiProjOut {
  const float* pb; const float* x; float* xres;
  __device__ void operator()(long long m, int col, float val) const {
    int b = (int)(m / (N_ * T_));
    int rem = (int)(m % (N_ * T_));
    int n = rem / T_, t = rem % T_;
    xres[m * C_ + col] = val + pb[col] + x[(((long long)b * C_ + col) * N_ + n) * T_ + t];
  }
};
struct EpiFfn1 {
  const float* b1; float* AG;
  __device__ void operator()(long long m, int col, float val) const {
    AG[m * 1024 + col] = val + b1[col];
  }
};
struct EpiFfn2 {
  const float* b2; const float* xres; float* out;
  __device__ void operator()(long long m, int col, float val) const {
    int b = (int)(m / (N_ * T_));
    int rem = (int)(m % (N_ * T_));
    int n = rem / T_, t = rem % T_;
    out[(((long long)b * C_ + col) * N_ + n) * T_ + t] = val + b2[col] + xres[m * C_ + col];
  }
};

template <int K, int NTILES, class Epi>
__global__ void __launch_bounds__(256) gemm_panel(const u16* __restrict__ A,
                                                  const u16* __restrict__ BT,
                                                  Epi epi) {
  constexpr int GROUPS = (NTILES + 7) / 8;
  __shared__ u16 sA[16 * K];
  int panel = blockIdx.x / GROUPS;
  int group = blockIdx.x % GROUPS;
  long long m0 = (long long)panel * 16;

  // ---- async-stage the 16xK A panel (contiguous) into LDS -----------------
  {
    const char* src = (const char*)(A + m0 * K);
    unsigned ldsBase = (unsigned)(size_t)(&sA[0]);
    constexpr int CHUNKS = (16 * K * 2) / 16;    // 16B chunks
    for (int i = threadIdx.x; i < CHUNKS; i += 256) {
      unsigned dst = ldsBase + (unsigned)(i * 16);
      const char* g = src + (long long)i * 16;
      asm volatile("global_load_async_to_lds_b128 %0, %1, off"
                   :: "v"(dst), "v"(g) : "memory");
    }
  }
  asm volatile("s_wait_asynccnt 0x0" ::: "memory");
  __syncthreads();

  int wave = threadIdx.x >> 5, lane = threadIdx.x & 31;
  int nt = group * 8 + wave;
  if (nt >= NTILES) return;                       // wave-uniform, after barrier
  int r = lane & 15, h = lane >> 4;
  const u16* Brow = BT + ((long long)nt * 16 + r) * K;   // lane's K-major column
  v8f acc = {};
#pragma unroll
  for (int k0 = 0; k0 < K; k0 += 32) {
    v16bf a, b;
#pragma unroll
    for (int e = 0; e < 16; ++e) a[e] = bfu(sA[r * K + k0 + kmapA(e, h)]);   // ds_load
#pragma unroll
    for (int e = 0; e < 16; ++e) b[e] = bfu(Brow[k0 + h * 16 + e]);          // b128 pair
    acc = wmma_bf16(a, b, acc);
  }
  int col = nt * 16 + r;
#pragma unroll
  for (int i = 0; i < 8; ++i) epi(m0 + i + 8 * h, col, acc[i]);
}

// ---------------------------------------------------------------------------
// Node-axis attention logits: per (b,t) slice, 768x768 = nq . nk^T
// ---------------------------------------------------------------------------
__global__ void gemm_node_logits(const u16* __restrict__ QK, const float* __restrict__ Bn,
                                 float* __restrict__ logits) {
  const long long perSlice = 48 * 48;
  int lane = threadIdx.x & 31, wave = threadIdx.x >> 5;
  long long tile = (long long)blockIdx.x * (blockDim.x >> 5) + wave;
  if (tile >= 96LL * perSlice) return;
  int s = (int)(tile / perSlice);
  int tt = (int)(tile % perSlice);
  int m0 = (tt / 48) * 16, n0 = (tt % 48) * 16;
  int b = s / 12, t = s % 12;
  int r = lane & 15, h = lane >> 4;
  const long long rowStride = 12 * 256;           // node i -> node i+1
  const u16* base = QK + (((long long)b * 768) * 12 + t) * 256;
  const u16* Aq = base + (long long)m0 * rowStride;        // nq at +0
  const u16* Bk = base + (long long)n0 * rowStride + 64;   // nk at +64
  v8f acc = {};
#pragma unroll
  for (int k0 = 0; k0 < 64; k0 += 32) {
    v16bf a = load_frag_a(Aq, rowStride, r, h, k0);
    v16bf b2 = load_frag_bt(Bk, rowStride, r, h, k0);
    acc = wmma_bf16(a, b2, acc);
  }
  long long obase = (long long)s * 768 * 768;
  int colj = n0 + r;
#pragma unroll
  for (int i = 0; i < 8; ++i) {
    int row = m0 + i + 8 * h;
    logits[obase + (long long)row * 768 + colj] =
        acc[i] * SCALE_ + Bn[row * 768 + colj];
  }
}

// ---------------------------------------------------------------------------
// Softmax * adj, stores TRANSPOSED bf16 attention: attnT[s][w][v] = attn[v][w]
// ---------------------------------------------------------------------------
__global__ void softmax_adj(const float* __restrict__ logits, const float* __restrict__ adj,
                            u16* __restrict__ attnT) {
  long long row = blockIdx.x;                     // 96*768 rows (v index)
  int s = (int)(row / 768);
  int i = (int)(row % 768);
  const float* lp = logits + row * 768;
  const float* ap = adj + row * 768;
  u16* opT = attnT + (long long)s * 768 * 768 + i;   // + w*768
  int tid = threadIdx.x;                          // 256
  __shared__ float red[256];
  float m = -1e30f;
#pragma unroll
  for (int k = 0; k < 3; ++k) m = fmaxf(m, lp[tid + k * 256]);
  red[tid] = m; __syncthreads();
  for (int st = 128; st > 0; st >>= 1) {
    if (tid < st) red[tid] = fmaxf(red[tid], red[tid + st]);
    __syncthreads();
  }
  float rowmax = red[0];
  __syncthreads();
  float e0[3], ssum = 0.f;
#pragma unroll
  for (int k = 0; k < 3; ++k) { e0[k] = __expf(lp[tid + k * 256] - rowmax); ssum += e0[k]; }
  red[tid] = ssum; __syncthreads();
  for (int st = 128; st > 0; st >>= 1) {
    if (tid < st) red[tid] += red[tid + st];
    __syncthreads();
  }
  float inv = 1.f / red[0];
#pragma unroll
  for (int k = 0; k < 3; ++k) {
    int j = tid + k * 256;
    opT[(long long)j * 768] = f2bf(e0[k] * inv * ap[j]);
  }
}

// ---------------------------------------------------------------------------
// hs = attn^T . V per slice -> H channels 64..127; both operands contiguous.
// ---------------------------------------------------------------------------
__global__ void gemm_hs(const u16* __restrict__ attnT, const u16* __restrict__ VbfT,
                        float* __restrict__ Hbuf) {
  const int perSlice = 48 * 4;
  int lane = threadIdx.x & 31, wave = threadIdx.x >> 5;
  long long tile = (long long)blockIdx.x * (blockDim.x >> 5) + wave;
  if (tile >= 96LL * perSlice) return;
  int s = (int)(tile / perSlice), tt = (int)(tile % perSlice);
  int m0 = (tt / 4) * 16, n0 = (tt % 4) * 16;
  int b = s / 12, t = s % 12;
  int r = lane & 15, h = lane >> 4;
  const u16* Ab = attnT + (long long)s * 768 * 768 + (long long)m0 * 768;  // [w][v]
  const u16* Bb = VbfT + ((long long)s * 64 + n0) * 768;                   // [c][v]
  v8f acc = {};
  for (int k0 = 0; k0 < 768; k0 += 32) {
    v16bf a = load_frag_a(Ab, 768, r, h, k0);
    v16bf bb = load_frag_bt(Bb, 768, r, h, k0);
    acc = wmma_bf16(a, bb, acc);
  }
  int c = n0 + r;
#pragma unroll
  for (int i = 0; i < 8; ++i) {
    int w = m0 + i + 8 * h;
    Hbuf[(((long long)b * 768 + w) * 12 + t) * 128 + 64 + c] = acc[i];
  }
}

// ---------------------------------------------------------------------------
// Time-axis attention (T=12, tiny) + ht -> H channels 0..63
// ---------------------------------------------------------------------------
__global__ void time_attn(const u16* __restrict__ QK, const u16* __restrict__ Vbf,
                          const float* __restrict__ Bt, float* __restrict__ Hbuf) {
  int bn = blockIdx.x;                            // 8*768
  long long b = bn / 768, n = bn % 768;
  __shared__ float q[12][64], k[12][64], att[12][12];
  int tid = threadIdx.x;                          // 256
  const u16* base = QK + ((b * 768 + n) * 12) * 256;
  for (int i = tid; i < 12 * 64; i += 256) {
    int tt = i / 64, d = i % 64;
    q[tt][d] = bf2f(base[tt * 256 + 128 + d]);    // tq
    k[tt][d] = bf2f(base[tt * 256 + 192 + d]);    // tk
  }
  __syncthreads();
  if (tid < 144) {
    int i = tid / 12, j = tid % 12;
    float s = 0.f;
#pragma unroll
    for (int d = 0; d < 64; ++d) s += q[i][d] * k[j][d];
    att[i][j] = s * SCALE_ + Bt[i * 12 + j];
  }
  __syncthreads();
  if (tid < 12) {
    float m = -1e30f;
    for (int j = 0; j < 12; ++j) m = fmaxf(m, att[tid][j]);
    float s = 0.f;
    for (int j = 0; j < 12; ++j) { float e = __expf(att[tid][j] - m); att[tid][j] = e; s += e; }
    float inv = 1.f / s;
    for (int j = 0; j < 12; ++j) att[tid][j] *= inv;
  }
  __syncthreads();
  const u16* vb = Vbf + ((b * 768 + n) * 12) * 64;
  for (int o = tid; o < 64 * 12; o += 256) {
    int c = o / 12, t = o % 12;
    float s = 0.f;
#pragma unroll
    for (int v = 0; v < 12; ++v) s += bf2f(vb[v * 64 + c]) * att[v][t];
    Hbuf[((b * 768 + n) * 12 + t) * 128 + c] = s;
  }
}

// y = h * fcv  (elementwise, bf16 out for proj GEMM)
__global__ void mul_hv(const float* __restrict__ H, const float* __restrict__ FCV,
                       u16* __restrict__ Y) {
  long long i = (long long)blockIdx.x * blockDim.x + threadIdx.x;
  if (i < (long long)P_ * 128) Y[i] = f2bf(H[i] * FCV[i]);
}

// gated GELU (exact): gated = a * 0.5*g*(1+erf(g/sqrt(2))) -> bf16
__global__ void gelu_gate(const float* __restrict__ AG, u16* __restrict__ G) {
  long long i = (long long)blockIdx.x * blockDim.x + threadIdx.x;
  if (i >= (long long)P_ * 512) return;
  long long p = i >> 9;
  int j = (int)(i & 511);
  float a = AG[p * 1024 + j];
  float g = AG[p * 1024 + 512 + j];
  float ge = 0.5f * g * (1.f + erff(g * 0.70710678118654752f));
  G[i] = f2bf(a * ge);
}

// ---------------------------------------------------------------------------
extern "C" void kernel_launch(void* const* d_in, const int* in_sizes, int n_in,
                              void* d_out, int out_size, void* d_ws, size_t ws_size,
                              hipStream_t stream) {
  (void)in_sizes; (void)n_in; (void)out_size; (void)ws_size;
  const float* x      = (const float*)d_in[0];
  const float* adj    = (const float*)d_in[4];
  const float* ln1w   = (const float*)d_in[5];
  const float* ln1b   = (const float*)d_in[6];
  const float* ln2w   = (const float*)d_in[7];
  const float* ln2b   = (const float*)d_in[8];
  const float* Wnq    = (const float*)d_in[9];
  const float* bnq    = (const float*)d_in[10];
  const float* Wnk    = (const float*)d_in[11];
  const float* bnk    = (const float*)d_in[12];
  const float* Wtq    = (const float*)d_in[13];
  const float* btq    = (const float*)d_in[14];
  const float* Wtk    = (const float*)d_in[15];
  const float* btk    = (const float*)d_in[16];
  const float* Bn     = (const float*)d_in[17];
  const float* Bt     = (const float*)d_in[18];
  const float* v_w    = (const float*)d_in[19];
  const float* v_b    = (const float*)d_in[20];
  const float* fcv_w  = (const float*)d_in[21];
  const float* fcv_b  = (const float*)d_in[22];
  const float* proj_w = (const float*)d_in[23];
  const float* proj_b = (const float*)d_in[24];
  const float* f1w    = (const float*)d_in[25];
  const float* f1b    = (const float*)d_in[26];
  const float* f2w    = (const float*)d_in[27];
  const float* f2b    = (const float*)d_in[28];
  float* out = (float*)d_out;

  // ---- workspace arena (256B carves, phase-based aliasing) ----------------
  char* ws = (char*)d_ws;
  auto carve = [&](size_t bytes) -> char* {
    char* p = ws;
    ws += (bytes + 255) & ~(size_t)255;
    return p;
  };
  float* bias448 = (float*)carve(448 * 4);
  u16*   Wp1T    = (u16*)carve((size_t)448 * 256 * 2);
  u16*   WprojT  = (u16*)carve((size_t)256 * 128 * 2);
  u16*   Wf1T    = (u16*)carve((size_t)1024 * 256 * 2);
  u16*   Wf2T    = (u16*)carve((size_t)256 * 512 * 2);
  u16*   X12     = (u16*)carve((size_t)P_ * 256 * 2);   // X1(LN1) then X2(LN2)
  u16*   QK      = (u16*)carve((size_t)P_ * 256 * 2);   // nq|nk|tq|tk
  u16*   Vbf     = (u16*)carve((size_t)P_ * 64 * 2);    // [p][64]
  u16*   VbfT    = (u16*)carve((size_t)P_ * 64 * 2);    // per-slice [c][n] K-major
  float* FCV     = (float*)carve((size_t)P_ * 128 * 4); // | contiguous pair,
  float* Hbuf    = (float*)carve((size_t)P_ * 128 * 4); // | reused as Gated
  u16*   Ybf     = (u16*)carve((size_t)P_ * 128 * 2);
  float* xres    = (float*)carve((size_t)P_ * 256 * 4);
  size_t logitsBytes = (size_t)96 * 768 * 768 * 4;
  char*  regA    = carve(logitsBytes + (size_t)96 * 768 * 768 * 2 + 256);
  float* logits  = (float*)regA;
  u16*   attnT   = (u16*)(regA + ((logitsBytes + 255) & ~(size_t)255));
  float* AG      = (float*)regA;   // FFN1 acts alias dead logits+attnT
  u16*   Gated   = (u16*)FCV;      // gated FFN acts alias dead FCV+Hbuf (75.5MB)

  pack_weights<<<1024, 256, 0, stream>>>(Wnq, Wnk, Wtq, Wtk, bnq, bnk, btq, btk,
                                         v_w, v_b, fcv_w, fcv_b, proj_w, f1w, f2w,
                                         Wp1T, WprojT, Wf1T, Wf2T, bias448);
  ln1_kernel<<<P_, 256, 0, stream>>>(x, ln1w, ln1b, X12);
  gemm_panel<256, 28, EpiProjAll><<<4608 * 4, 256, 0, stream>>>(
      X12, Wp1T, EpiProjAll{bias448, QK, Vbf, VbfT, FCV});
  gemm_node_logits<<<27648, 256, 0, stream>>>(QK, Bn, logits);   // 96*48*48/8
  softmax_adj<<<96 * 768, 256, 0, stream>>>(logits, adj, attnT);
  gemm_hs<<<2304, 256, 0, stream>>>(attnT, VbfT, Hbuf);          // 96*48*4/8
  time_attn<<<B_ * N_, 256, 0, stream>>>(QK, Vbf, Bt, Hbuf);
  mul_hv<<<(P_ * 128) / 256, 256, 0, stream>>>(Hbuf, FCV, Ybf);
  gemm_panel<128, 16, EpiProjOut><<<4608 * 2, 256, 0, stream>>>(
      Ybf, WprojT, EpiProjOut{proj_b, x, xres});
  ln2_kernel<<<P_, 256, 0, stream>>>(xres, ln2w, ln2b, X12);
  gemm_panel<256, 64, EpiFfn1><<<4608 * 8, 256, 0, stream>>>(
      X12, Wf1T, EpiFfn1{f1b, AG});
  gelu_gate<<<(P_ * 512) / 256, 256, 0, stream>>>(AG, Gated);
  gemm_panel<512, 16, EpiFfn2><<<4608 * 2, 256, 0, stream>>>(
      Gated, Wf2T, EpiFfn2{f2b, xres, out});
}
